// GNNRegressor_17162689315160
// MI455X (gfx1250) — compile-verified, matrix operand order checked by reading, not run on hardware
//
#include <hip/hip_runtime.h>
#include <stdint.h>

#define NNODES 50000
#define NEDGES 800000
#define HDIM   128
#define RT     5              // row tiles per wave -> 80-row M-tile per block
#define MROWS  (16 * RT)      // 80; 50000 / 80 = 625 exactly

typedef __attribute__((ext_vector_type(16))) _Float16 v16h;
typedef __attribute__((ext_vector_type(8)))  _Float16 v8h;
typedef __attribute__((ext_vector_type(8)))  float    v8f;

// ---------------- degree / norm ----------------
__global__ void k_fill(float* __restrict__ p, float val, int n) {
    int i = blockIdx.x * blockDim.x + threadIdx.x;
    if (i < n) p[i] = val;
}

__global__ void k_deg_edges(const long long* __restrict__ ei, float* __restrict__ deg) {
    int e = blockIdx.x * blockDim.x + threadIdx.x;
    if (e < NEDGES) atomicAdd(&deg[(int)ei[NEDGES + e]], 1.0f);
}

__global__ void k_dinv(const float* __restrict__ deg, float* __restrict__ dinv) {
    int i = blockIdx.x * blockDim.x + threadIdx.x;
    if (i < NNODES) dinv[i] = rsqrtf(deg[i]);   // deg >= 1 always (self loop)
}

// ---- weight transpose+convert: WT[n*128+k] = (f16) W[k*Ncols+n] ----
__global__ void k_wt(const float* __restrict__ W, _Float16* __restrict__ WT, int Ncols) {
    int idx = blockIdx.x * blockDim.x + threadIdx.x;
    if (idx >= Ncols * 128) return;
    int n = idx >> 7, k = idx & 127;
    WT[idx] = (_Float16)W[k * Ncols + n];
}

// ---------------- WMMA GEMM with fused epilogue ----------------
// C = A[M x 128] * B, B supplied transposed as BT[Ncols x 128] f16.
// grid.x = M/80 ; blockDim.x = 32 * (Ncols/16). Each wave: one 16-col tile,
// 5 row tiles -> 20 v_wmma per wave, B fragment reused 5x.
// Conv mode (dinv != null): xw = A*B ; agg = bias + dinv^2 * xw   (self-loop init)
// Head mode (dinv == null): xw = relu(A*B + bias)
__global__ void k_gemm(const float* __restrict__ A, const _Float16* __restrict__ BT,
                       const float* __restrict__ bias, const float* __restrict__ dinv,
                       float* __restrict__ xw, float* __restrict__ agg, int Ncols) {
    __shared__ __align__(16) _Float16 lA[MROWS * 128];   // 20 KB
    __shared__ float lD[MROWS];
    const int tid  = threadIdx.x;
    const int row0 = blockIdx.x * MROWS;

    // stage + convert A tile (80 rows x 128 K) to f16 in LDS
    for (int i = tid; i < MROWS * 128; i += blockDim.x) {
        int r = i >> 7, k = i & 127;
        lA[i] = (_Float16)A[(size_t)(row0 + r) * 128 + k];
    }
    if (dinv && tid < MROWS) lD[tid] = dinv[row0 + tid];
    __syncthreads();

    const int lane = tid & 31;
    const int wv   = tid >> 5;          // wave id -> 16-col tile
    const int nr   = lane & 15;         // A row within tile / B column within tile
    const int hi   = lane >> 4;         // K-half selector per WMMA lane layout
    const int col  = (wv << 4) + nr;    // global output column
    const _Float16* bcol = BT + (size_t)col * 128;

    v8f c[RT];
#pragma unroll
    for (int t = 0; t < RT; ++t) c[t] = (v8f){};

#pragma unroll
    for (int kb = 0; kb < 128; kb += 32) {
        // B lane fragment: K = kb + hi*8 .. +7  and  kb + 16 + hi*8 .. +7
        v8h b0 = *(const v8h*)(bcol + kb + hi * 8);
        v8h b1 = *(const v8h*)(bcol + kb + 16 + hi * 8);
        v16h b;
#pragma unroll
        for (int t = 0; t < 8; ++t) { b[t] = b0[t]; b[t + 8] = b1[t]; }
#pragma unroll
        for (int rt = 0; rt < RT; ++rt) {
            const _Float16* arow = &lA[(rt * 16 + nr) * 128 + kb + hi * 8];
            v8h a0 = *(const v8h*)(arow);
            v8h a1 = *(const v8h*)(arow + 16);
            v16h a;
#pragma unroll
            for (int t = 0; t < 8; ++t) { a[t] = a0[t]; a[t + 8] = a1[t]; }
            c[rt] = __builtin_amdgcn_wmma_f32_16x16x32_f16(false, a, false, b,
                                                           (short)0, c[rt], false, false);
        }
    }

    const float bv = bias[col];
#pragma unroll
    for (int rt = 0; rt < RT; ++rt) {
#pragma unroll
        for (int r = 0; r < 8; ++r) {   // VGPR r: lanes0-15 M=r, lanes16-31 M=r+8
            int lr  = rt * 16 + r + hi * 8;          // local row
            size_t gi = (size_t)(row0 + lr) * Ncols + col;
            float v = c[rt][r];
            if (dinv) {                              // conv layer epilogue
                float di = lD[lr];
                xw[gi]  = v;
                agg[gi] = bv + di * di * v;
            } else {                                 // head: bias + relu
                xw[gi] = fmaxf(v + bv, 0.0f);
            }
        }
    }
}

// wave per edge: agg[dst] += dinv[src]*dinv[dst] * xw[src]
__global__ void k_agg_edges(const long long* __restrict__ ei, const float* __restrict__ dinv,
                            const float* __restrict__ xw, float* __restrict__ agg) {
    int e = blockIdx.x * 8 + (threadIdx.x >> 5);
    if (e >= NEDGES) return;
    int lane = threadIdx.x & 31;
    int s = (int)ei[e];
    int d = (int)ei[NEDGES + e];
    float w = dinv[s] * dinv[d];
    float4 v = ((const float4*)(xw + (size_t)s * HDIM))[lane];
    float* ar = agg + (size_t)d * HDIM + lane * 4;
    atomicAdd(ar + 0, w * v.x);
    atomicAdd(ar + 1, w * v.y);
    atomicAdd(ar + 2, w * v.z);
    atomicAdd(ar + 3, w * v.w);
}

// ---------------- BN(eval)+ReLU in place, residual accumulate ----------------
__global__ void k_bn_relu_sum(float* __restrict__ agg,
                              const float* __restrict__ g, const float* __restrict__ be,
                              const float* __restrict__ m, const float* __restrict__ v,
                              float* __restrict__ hsum, int first) {
    int idx = blockIdx.x * blockDim.x + threadIdx.x;
    if (idx >= NNODES * HDIM) return;
    int c = idx & 127;
    float sc = g[c] * rsqrtf(v[c] + 1e-5f);
    float h  = (agg[idx] - m[c]) * sc + be[c];
    h = fmaxf(h, 0.0f);
    agg[idx]  = h;
    hsum[idx] = first ? h : hsum[idx] + h;
}

// ---------------- final head layer: out[i] = lb2 + hd[i] . lw2 ----------------
__global__ void k_head2(const float* __restrict__ hd, const float* __restrict__ lw2,
                        const float* __restrict__ lb2, float* __restrict__ out) {
    int i = blockIdx.x * blockDim.x + threadIdx.x;
    if (i >= NNODES) return;
    const float4* r = (const float4*)(hd + (size_t)i * 64);
    const float4* w = (const float4*)lw2;
    float acc = lb2[0];
#pragma unroll
    for (int j = 0; j < 16; ++j) {
        float4 a = r[j], b = w[j];
        acc += a.x * b.x + a.y * b.y + a.z * b.z + a.w * b.w;
    }
    out[i] = acc;
}

extern "C" void kernel_launch(void* const* d_in, const int* in_sizes, int n_in,
                              void* d_out, int out_size, void* d_ws, size_t ws_size,
                              hipStream_t stream) {
    const float*     x   = (const float*)d_in[0];
    const long long* ei  = (const long long*)d_in[1];
    const float *W1 = (const float*)d_in[2],  *b1 = (const float*)d_in[3];
    const float *W2 = (const float*)d_in[4],  *b2 = (const float*)d_in[5];
    const float *W3 = (const float*)d_in[6],  *b3 = (const float*)d_in[7];
    const float *g1 = (const float*)d_in[8],  *be1 = (const float*)d_in[9];
    const float *m1 = (const float*)d_in[10], *v1  = (const float*)d_in[11];
    const float *g2 = (const float*)d_in[12], *be2 = (const float*)d_in[13];
    const float *m2 = (const float*)d_in[14], *v2  = (const float*)d_in[15];
    const float *g3 = (const float*)d_in[16], *be3 = (const float*)d_in[17];
    const float *m3 = (const float*)d_in[18], *v3  = (const float*)d_in[19];
    const float *lw1 = (const float*)d_in[20], *lb1 = (const float*)d_in[21];
    const float *lw2 = (const float*)d_in[22], *lb2 = (const float*)d_in[23];
    float* out = (float*)d_out;

    // carve workspace
    uintptr_t p = (uintptr_t)d_ws;
    auto carve = [&](size_t bytes) -> void* {
        p = (p + 255) & ~(uintptr_t)255;
        void* r = (void*)p;
        p += bytes;
        return r;
    };
    float* deg   = (float*)carve((size_t)NNODES * 4);
    float* dinv  = (float*)carve((size_t)NNODES * 4);
    float* xw    = (float*)carve((size_t)NNODES * HDIM * 4);
    float* agg   = (float*)carve((size_t)NNODES * HDIM * 4);
    float* hsum  = (float*)carve((size_t)NNODES * HDIM * 4);
    float* hd    = (float*)carve((size_t)NNODES * 64 * 4);
    _Float16* WT1  = (_Float16*)carve((size_t)HDIM * HDIM * 2);
    _Float16* WT2  = (_Float16*)carve((size_t)HDIM * HDIM * 2);
    _Float16* WT3  = (_Float16*)carve((size_t)HDIM * HDIM * 2);
    _Float16* LWT1 = (_Float16*)carve((size_t)64 * HDIM * 2);

    const int nb_nodes = (NNODES + 255) / 256;
    const int nb_edges = (NEDGES + 255) / 256;
    const int nb_elems = (NNODES * HDIM) / 256;       // 25000
    const int mTiles   = NNODES / MROWS;              // 625 (exact)
    const int edgeBlocks = (NEDGES + 7) / 8;

    // ---- degrees & symmetric norm factors (shared by all 3 conv layers) ----
    k_fill<<<nb_nodes, 256, 0, stream>>>(deg, 1.0f, NNODES);
    k_deg_edges<<<nb_edges, 256, 0, stream>>>(ei, deg);
    k_dinv<<<nb_nodes, 256, 0, stream>>>(deg, dinv);

    // ---- weight conversion to transposed f16 ----
    k_wt<<<(HDIM * 128 + 255) / 256, 256, 0, stream>>>(W1, WT1, HDIM);
    k_wt<<<(HDIM * 128 + 255) / 256, 256, 0, stream>>>(W2, WT2, HDIM);
    k_wt<<<(HDIM * 128 + 255) / 256, 256, 0, stream>>>(W3, WT3, HDIM);
    k_wt<<<(64 * 128 + 255) / 256, 256, 0, stream>>>(lw1, LWT1, 64);

    // ---- layer 1: A = x ----
    k_gemm<<<mTiles, 256, 0, stream>>>(x, WT1, b1, dinv, xw, agg, HDIM);
    k_agg_edges<<<edgeBlocks, 256, 0, stream>>>(ei, dinv, xw, agg);
    k_bn_relu_sum<<<nb_elems, 256, 0, stream>>>(agg, g1, be1, m1, v1, hsum, 1);

    // ---- layer 2: A = h1 (in agg) ----
    k_gemm<<<mTiles, 256, 0, stream>>>(agg, WT2, b2, dinv, xw, agg, HDIM);
    k_agg_edges<<<edgeBlocks, 256, 0, stream>>>(ei, dinv, xw, agg);
    k_bn_relu_sum<<<nb_elems, 256, 0, stream>>>(agg, g2, be2, m2, v2, hsum, 0);

    // ---- layer 3: A = h2 (in agg) ----
    k_gemm<<<mTiles, 256, 0, stream>>>(agg, WT3, b3, dinv, xw, agg, HDIM);
    k_agg_edges<<<edgeBlocks, 256, 0, stream>>>(ei, dinv, xw, agg);
    k_bn_relu_sum<<<nb_elems, 256, 0, stream>>>(agg, g3, be3, m3, v3, hsum, 0);

    // ---- head: hd = relu(hsum @ lw1 + lb1) via WMMA (Ncols=64 -> 4 waves) ----
    k_gemm<<<mTiles, 128, 0, stream>>>(hsum, LWT1, lb1, nullptr, hd, nullptr, 64);
    // ---- final: out = hd @ lw2 + lb2 ----
    k_head2<<<nb_nodes, 256, 0, stream>>>(hd, lw2, lb2, out);
}
